// LiquidNeuralNetwork_23545010716813
// MI455X (gfx1250) — compile-verified, matrix operand order checked by reading, not run on hardware
//
#include <hip/hip_runtime.h>
#include <hip/hip_bf16.h>

typedef __attribute__((ext_vector_type(16))) _Float16 v16h;
typedef __attribute__((ext_vector_type(8)))  float    v8f;

namespace {

constexpr int H     = 128;   // hidden size
constexpr int BATCH = 512;
constexpr int SEQ   = 512;
constexpr int FSEQ  = 64;
constexpr int FCTX  = 32;
constexpr int IN0   = 96;    // FSEQ + FCTX
constexpr int ROWS  = 16;    // batch rows per workgroup (one WMMA M tile)
constexpr int SW96  = 104;   // padded LDS row stride (halves), K=96  (13*16B rows)
constexpr int SW128 = 136;   // padded LDS row stride (halves), K=128 (17*16B rows)
constexpr int XBUF  = ROWS * SW96;   // one x-staging buffer (halves)

constexpr size_t SMEM_HALVES =
    2 * (size_t)H * SW96  +   // Win0, Wg0_x
    2 * (size_t)H * SW128 +   // Win1, Wg1_x
    2 * (size_t)XBUF      +   // x16 : double-buffered layer-0 input
    (size_t)ROWS * SW128  +   // xh1 : layer-1 input (h0)
    2 * (size_t)ROWS * SW128; // aBuf: double-buffered hh staging
constexpr size_t SMEM_BYTES =
    SMEM_HALVES * 2 + (size_t)(2 * H + 2 * H + ROWS * H) * sizeof(float);
static_assert(SMEM_BYTES <= 160u * 1024u, "comfortably under WGP LDS");

__device__ __forceinline__ float fast_tanh(float x) {
#if __has_builtin(__builtin_amdgcn_tanhf)
  return __builtin_amdgcn_tanhf(x);        // v_tanh_f32 (TRANS)
#else
  return tanhf(x);
#endif
}
__device__ __forceinline__ float fast_sigmoid(float x) {
#if __has_builtin(__builtin_amdgcn_exp2f) && __has_builtin(__builtin_amdgcn_rcpf)
  float e = __builtin_amdgcn_exp2f(x * -1.44269504088896340736f); // exp(-x)
  return __builtin_amdgcn_rcpf(1.0f + e);                        // v_rcp_f32
#else
  return 1.0f / (1.0f + expf(-x));
#endif
}

__device__ __forceinline__ v16h load_frag16(const _Float16* p) {
  v16h r;
  ((uint4*)&r)[0] = ((const uint4*)p)[0];   // 16B-aligned ds_load_b128 pair
  ((uint4*)&r)[1] = ((const uint4*)p)[1];
  return r;
}

// B fragment straight from global fp32 weights (one-time): lane holds 16
// consecutive K values of its output column. W row-major [out][ldw].
__device__ __forceinline__ v16h load_wfrag_global(const float* __restrict__ W,
                                                  int ldw, int o, int k0) {
  v16h r;
  #pragma unroll
  for (int i = 0; i < 16; ++i) r[i] = (_Float16)W[(size_t)o * ldw + k0 + i];
  return r;
}

__device__ __forceinline__ v8f wmma_f16(const v16h& a, const v16h& b, const v8f& c) {
  return __builtin_amdgcn_wmma_f32_16x16x32_f16(false, a, false, b,
                                                (short)0, c, false, false);
}

// Two 16x16 tiles sharing the A operand, B from LDS. All fragments prefetched
// into registers first so DS loads drain with partial waits.
__device__ __forceinline__ void gemm_pair_lds(
    const _Float16* A, int strideA, int kChunks,
    const _Float16* W0, const _Float16* W1, int strideW,
    int lane, int colBase, v8f& acc0, v8f& acc1) {
  const int rA  = lane & 15;
  const int kHi = (lane >> 4) * 16;
  const int oW  = colBase + (lane & 15);
  v16h a[4], b0[4], b1[4];
  #pragma unroll
  for (int kc = 0; kc < 4; ++kc) {
    if (kc < kChunks) {
      const int k0 = kc * 32 + kHi;
      a[kc]  = load_frag16(A  + rA * strideA + k0);
      b0[kc] = load_frag16(W0 + oW * strideW + k0);
      b1[kc] = load_frag16(W1 + oW * strideW + k0);
    }
  }
  #pragma unroll
  for (int kc = 0; kc < 4; ++kc) {
    if (kc < kChunks) {
      acc0 = wmma_f16(a[kc], b0[kc], acc0);
      acc1 = wmma_f16(a[kc], b1[kc], acc1);
    }
  }
}

// Two 16x16 tiles sharing the A operand, B register-resident (per-stage GEMMs).
__device__ __forceinline__ void gemm_pair_reg(
    const _Float16* A, int strideA,
    const v16h* b0, const v16h* b1,
    int lane, v8f& acc0, v8f& acc1) {
  const int rA  = lane & 15;
  const int kHi = (lane >> 4) * 16;
  v16h a[4];
  #pragma unroll
  for (int kc = 0; kc < 4; ++kc)
    a[kc] = load_frag16(A + rA * strideA + kc * 32 + kHi);
  #pragma unroll
  for (int kc = 0; kc < 4; ++kc) {
    acc0 = wmma_f16(a[kc], b0[kc], acc0);
    acc1 = wmma_f16(a[kc], b1[kc], acc1);
  }
}

// C/D fragment (v8f): VGPR r -> row r + 8*(lane>>4), col = lane&15 (ISA 7.12.2)
__device__ __forceinline__ void store_frag_f16(_Float16* dst, int stride,
                                               int colBase, int lane, const v8f& v) {
  const int col = colBase + (lane & 15);
  const int r0  = (lane >> 4) * 8;
  #pragma unroll
  for (int r = 0; r < 8; ++r) dst[(r0 + r) * stride + col] = (_Float16)v[r];
}

// RK4 stage loop (U, Zx precomputed by caller). One barrier per stage via
// double-buffered hh staging.
__device__ __forceinline__ void ltc_stages(
    const v8f& U, const v8f& Zx,
    const v16h* brec, const v16h* bgh,
    _Float16* aBuf, float invtau,
    int lane, int colBase, v8f& h)
{
  v8f ksum = {};
  v8f kprev = {};
  const float cs[4] = {0.0f, 0.5f, 0.5f, 1.0f};
  const float ws[4] = {1.0f, 2.0f, 2.0f, 1.0f};
  #pragma unroll
  for (int s = 0; s < 4; ++s) {
    v8f hh;
    #pragma unroll
    for (int r = 0; r < 8; ++r) hh[r] = h[r] + cs[s] * kprev[r];
    _Float16* a16 = aBuf + (s & 1) * (ROWS * SW128);
    store_frag_f16(a16, SW128, colBase, lane, hh);
    __syncthreads();                      // single barrier: write -> read
    v8f G = {}, Zh = {};
    gemm_pair_reg(a16, SW128, brec, bgh, lane, G, Zh);
    #pragma unroll
    for (int r = 0; r < 8; ++r) {
      float g = fast_sigmoid(fast_tanh(Zx[r] + Zh[r]));
      float k = -hh[r] * invtau + U[r] + g * G[r];
      ksum[r] += ws[s] * k;
      kprev[r] = k;
    }
  }
  #pragma unroll
  for (int r = 0; r < 8; ++r) h[r] = fast_tanh(h[r] + ksum[r] * (1.0f / 6.0f));
}

} // namespace

extern "C" __global__ void __launch_bounds__(256)
ltc_rk4_kernel(const float* __restrict__ seq,  const float* __restrict__ ctx,
               const float* __restrict__ tau0, const float* __restrict__ Win0,
               const float* __restrict__ Wrec0,const float* __restrict__ Wg0,
               const float* __restrict__ bg0,
               const float* __restrict__ tau1, const float* __restrict__ Win1,
               const float* __restrict__ Wrec1,const float* __restrict__ Wg1,
               const float* __restrict__ bg1,
               const float* __restrict__ W1,   const float* __restrict__ b1,
               const float* __restrict__ W2,   const float* __restrict__ b2,
               float* __restrict__ out)
{
  extern __shared__ __align__(16) char smem_raw[];
  _Float16* w_in0 = (_Float16*)smem_raw;       // [128][104]
  _Float16* w_gx0 = w_in0 + H * SW96;          // [128][104]
  _Float16* w_in1 = w_gx0 + H * SW96;          // [128][136]
  _Float16* w_gx1 = w_in1 + H * SW128;         // [128][136]
  _Float16* x16   = w_gx1 + H * SW128;         // 2x[16][104] layer-0 input
  _Float16* xh1   = x16   + 2 * XBUF;          // [16][136] layer-1 input (h0)
  _Float16* aBuf  = xh1   + ROWS * SW128;      // 2x[16][136] hh staging
  float* invtau = (float*)(aBuf + 2 * ROWS * SW128); // [2][128]
  float* bgv    = invtau + 2 * H;                    // [2][128]
  float* hcls   = bgv + 2 * H;                       // [16][128]

  const int tid = threadIdx.x;
  const int rowBase = blockIdx.x * ROWS;    // this WG's batch rows

  // ---- per-step weights (Win, Wg_x) -> fp16 LDS (once) ----
  for (int i = tid; i < H * IN0; i += 256) {
    int o = i / IN0, k = i % IN0;
    w_in0[o * SW96 + k] = (_Float16)Win0[i];
    w_gx0[o * SW96 + k] = (_Float16)Wg0[o * (IN0 + H) + k];
  }
  for (int i = tid; i < H * H; i += 256) {
    int o = i >> 7, k = i & 127;
    w_in1[o * SW128 + k] = (_Float16)Win1[i];
    w_gx1[o * SW128 + k] = (_Float16)Wg1[o * (2 * H) + k];
  }
  for (int i = tid; i < H; i += 256) {
    invtau[i]     = 1.0f / (log1pf(expf(tau0[i])) + 1.0f);  // 1/(softplus+1)
    invtau[H + i] = 1.0f / (log1pf(expf(tau1[i])) + 1.0f);
    bgv[i]     = bg0[i];
    bgv[H + i] = bg1[i];
  }
  // ---- x staging prologue: ctx half is t-invariant -> both buffers; seq[0] ----
  for (int i = tid; i < ROWS * FCTX; i += 256) {
    int r = i >> 5, c = i & 31;
    _Float16 v = (_Float16)ctx[(size_t)(rowBase + r) * FCTX + c];
    x16[r * SW96 + FSEQ + c]        = v;
    x16[XBUF + r * SW96 + FSEQ + c] = v;
  }
  for (int i = tid; i < ROWS * FSEQ; i += 256) {
    int r = i >> 6, c = i & 63;
    x16[r * SW96 + c] = (_Float16)seq[((size_t)(rowBase + r) * SEQ) * FSEQ + c];
  }

  const int lane    = tid & 31;
  const int colBase = (tid >> 5) * 16;      // wave w owns H columns [16w, 16w+16)

  // ---- per-stage weights (Wrec, Wg_h) -> register-resident B fragments ----
  v16h brec0[4], bgh0[4], brec1[4], bgh1[4];
  {
    const int kHi = (lane >> 4) * 16;
    const int oW  = colBase + (lane & 15);
    #pragma unroll
    for (int kc = 0; kc < 4; ++kc) {
      const int k0 = kc * 32 + kHi;
      brec0[kc] = load_wfrag_global(Wrec0, H,       oW, k0);
      bgh0[kc]  = load_wfrag_global(Wg0,   IN0 + H, oW, IN0 + k0);
      brec1[kc] = load_wfrag_global(Wrec1, H,       oW, k0);
      bgh1[kc]  = load_wfrag_global(Wg1,   2 * H,   oW, H + k0);
    }
  }
  __syncthreads();

  const int col   = colBase + (lane & 15);
  const float it0 = invtau[col], it1 = invtau[H + col];
  const float bgs0 = bgv[col],   bgs1 = bgv[H + col];

  v8f h0 = {};
  v8f h1 = {};
  for (int t = 0; t < SEQ; ++t) {
    const _Float16* xA = x16 + (t & 1) * XBUF;
    // prefetch seq[t+1] into registers: global loads don't touch DScnt, so
    // they fly under all the stage barriers of this step
    float pre[4];
    const bool more = (t + 1 < SEQ);
    if (more) {
      #pragma unroll
      for (int j = 0; j < 4; ++j) {
        int i = tid + j * 256;
        int r = i >> 6, c = i & 63;
        pre[j] = seq[((size_t)(rowBase + r) * SEQ + (t + 1)) * FSEQ + c];
      }
    }
    // ---- layer 0 ----
    v8f U0 = {}, Zx0 = {};
    gemm_pair_lds(xA, SW96, 3, w_in0, w_gx0, SW96, lane, colBase, U0, Zx0);
    #pragma unroll
    for (int r = 0; r < 8; ++r) Zx0[r] += bgs0;
    ltc_stages(U0, Zx0, brec0, bgh0, aBuf, it0, lane, colBase, h0);
    // drop prefetched seq[t+1] into the other x buffer; the xh1 barrier below
    // orders these stores before their reads next step
    if (more) {
      _Float16* xN = x16 + ((t + 1) & 1) * XBUF;
      #pragma unroll
      for (int j = 0; j < 4; ++j) {
        int i = tid + j * 256;
        int r = i >> 6, c = i & 63;
        xN[r * SW96 + c] = (_Float16)pre[j];
      }
    }
    store_frag_f16(xh1, SW128, colBase, lane, h0);   // h0 feeds layer 1
    __syncthreads();
    // ---- layer 1 ----
    v8f U1 = {}, Zx1 = {};
    gemm_pair_lds(xh1, SW128, 4, w_in1, w_gx1, SW128, lane, colBase, U1, Zx1);
    #pragma unroll
    for (int r = 0; r < 8; ++r) Zx1[r] += bgs1;
    ltc_stages(U1, Zx1, brec1, bgh1, aBuf, it1, lane, colBase, h1);
  }

  // ---- tiny classifier head: sigmoid(relu(h1 @ W1^T + b1) @ W2^T + b2) ----
  {
    const int r0 = (lane >> 4) * 8;
    #pragma unroll
    for (int r = 0; r < 8; ++r) hcls[(r0 + r) * H + col] = h1[r];
  }
  __syncthreads();
  if (tid < ROWS) {
    float acc = b2[0];
    for (int j = 0; j < H / 2; ++j) {
      float a = b1[j];
      for (int k = 0; k < H; ++k) a += hcls[tid * H + k] * W1[j * H + k];
      acc += fmaxf(a, 0.0f) * W2[j];
    }
    out[rowBase + tid] = fast_sigmoid(acc);
  }
}

extern "C" void kernel_launch(void* const* d_in, const int* in_sizes, int n_in,
                              void* d_out, int out_size, void* d_ws, size_t ws_size,
                              hipStream_t stream) {
  (void)in_sizes; (void)n_in; (void)d_ws; (void)ws_size; (void)out_size;
  hipFuncSetAttribute(reinterpret_cast<const void*>(ltc_rk4_kernel),
                      hipFuncAttributeMaxDynamicSharedMemorySize,
                      (int)SMEM_BYTES);
  ltc_rk4_kernel<<<BATCH / ROWS, 256, SMEM_BYTES, stream>>>(
      (const float*)d_in[0],  (const float*)d_in[1],
      (const float*)d_in[2],  (const float*)d_in[3],  (const float*)d_in[4],
      (const float*)d_in[5],  (const float*)d_in[6],
      (const float*)d_in[7],  (const float*)d_in[8],  (const float*)d_in[9],
      (const float*)d_in[10], (const float*)d_in[11],
      (const float*)d_in[12], (const float*)d_in[13],
      (const float*)d_in[14], (const float*)d_in[15],
      (float*)d_out);
}
